// LSPD_43404939493531
// MI455X (gfx1250) — compile-verified, matrix operand order checked by reading, not run on hardware
//
#include <hip/hip_runtime.h>

#define Bn 512
#define Tn 512
#define Dn 128
#define EMBn 32
#define Un 32
#define NCn 11

typedef __attribute__((ext_vector_type(16))) __bf16 v16bf;
typedef __attribute__((ext_vector_type(8)))  float  v8f;

struct alignas(16) U4 { unsigned int a, b, c, d; };
union FragU { unsigned int u[8]; U4 q[2]; v16bf v; };

static __device__ __forceinline__ unsigned short f2bf(float f) {
  unsigned int u = __float_as_uint(f);
  return (unsigned short)((u + 0x7FFFu + ((u >> 16) & 1u)) >> 16);
}

// Fast sigmoid: v_exp_f32 + raw v_rcp_f32 (avoids IEEE div expansion on the
// serial recurrence's critical path).
static __device__ __forceinline__ float fsigmoid(float x) {
  return __builtin_amdgcn_rcpf(1.0f + __expf(-x));
}

// Build a 32x16 bf16 B-operand fragment from a row-major f32 matrix (ld = #cols).
// Layout mirrors the documented 16-bit A layout: lane n holds column col0+(lane&15),
// lane-half selects K banks {0..7,16..23} (half 0) / {8..15,24..31} (half 1).
static __device__ __forceinline__ v16bf make_bfrag(const float* __restrict__ W, int ld,
                                                   int col0, int lane) {
  int n = col0 + (lane & 15);
  int half = lane >> 4;
  FragU f;
#pragma unroll
  for (int g = 0; g < 8; ++g) {
    int kb = (g < 4) ? (2 * g + 8 * half) : (16 + 2 * (g - 4) + 8 * half);
    f.u[g] = (unsigned int)f2bf(W[kb * ld + n]) |
             ((unsigned int)f2bf(W[(kb + 1) * ld + n]) << 16);
  }
  return f.v;
}

// ---------------------------------------------------------------------------
// K1: argmax over D=128 + embedding gather -> X (T,B,32) bf16, time-major
// ---------------------------------------------------------------------------
__global__ void argmax_embed(const float* __restrict__ x, const float* __restrict__ emb,
                             unsigned short* __restrict__ X) {
  int lane = threadIdx.x & 31;
  int wid = (blockIdx.x * blockDim.x + threadIdx.x) >> 5;
  if (wid >= Bn * Tn) return;
  int b = wid / Tn, t = wid % Tn;
  const float* row = x + ((size_t)b * Tn + t) * Dn;
  float bestv = -3.4e38f;
  int besti = 0;
#pragma unroll
  for (int k = 0; k < 4; ++k) {
    int i = lane + 32 * k;
    float v = row[i];
    if (v > bestv || (v == bestv && i < besti)) { bestv = v; besti = i; }
  }
#pragma unroll
  for (int off = 16; off > 0; off >>= 1) {
    float ov = __shfl_xor(bestv, off, 32);
    int   oi = __shfl_xor(besti, off, 32);
    if (ov > bestv || (ov == bestv && oi < besti)) { bestv = ov; besti = oi; }
  }
  X[((size_t)t * Bn + b) * 32 + lane] = f2bf(emb[(size_t)besti * EMBn + lane]);
}

// ---------------------------------------------------------------------------
// zero helper
// ---------------------------------------------------------------------------
__global__ void zero_f32(float* __restrict__ p, int n) {
  int i = blockIdx.x * blockDim.x + threadIdx.x;
  if (i < n) p[i] = 0.0f;
}

// ---------------------------------------------------------------------------
// K3: sequential bidirectional GRU. 1 wave owns 16 batch rows for one direction,
// marches over 511 time steps. 12 v_wmma_f32_16x16x32_bf16 per step.
// ---------------------------------------------------------------------------
__global__ __launch_bounds__(32) void gru_kernel(
    const unsigned short* __restrict__ X,  // (T,B,32) bf16
    float* __restrict__ outL, float* __restrict__ outR,  // (T,B,32) f32 each
    const float* __restrict__ WL, const float* __restrict__ UL, const float* __restrict__ bL,
    const float* __restrict__ WR, const float* __restrict__ UR, const float* __restrict__ bR) {
  const int lane = threadIdx.x;
  const int dir = blockIdx.y;
  const int brow0 = blockIdx.x * 16;
  const int n = lane & 15, half = lane >> 4;

  const float* W  = dir ? WR : WL;
  const float* U  = dir ? UR : UL;
  const float* bI = (dir ? bR : bL);        // input bias b[0]
  const float* bH = (dir ? bR : bL) + 96;   // recurrent bias b[1]
  float* out = dir ? outR : outL;

  // Resident weight fragments (K=32 -> one WMMA per GEMM per N-tile).
  v16bf Wf[6], Uf[6];
#pragma unroll
  for (int j = 0; j < 6; ++j) {
    Wf[j] = make_bfrag(W, 96, 16 * j, lane);
    Uf[j] = make_bfrag(U, 96, 16 * j, lane);
  }
  // Bias folded into accumulator init (per-column constant across C rows).
  float biasA[6], biasR2[2];
#pragma unroll
  for (int j = 0; j < 6; ++j) biasA[j] = bI[16 * j + n] + (j < 4 ? bH[16 * j + n] : 0.0f);
#pragma unroll
  for (int j = 0; j < 2; ++j) biasR2[j] = bH[64 + 16 * j + n];

  __shared__ __align__(16) unsigned short hs[16][32];

  float hstate[2][8];
#pragma unroll
  for (int j = 0; j < 2; ++j)
#pragma unroll
    for (int r = 0; r < 8; ++r) hstate[j][r] = 0.0f;

  FragU hz;
#pragma unroll
  for (int g = 0; g < 8; ++g) hz.u[g] = 0u;
  v16bf hfrag = hz.v;

  for (int s = 0; s < Tn - 1; ++s) {
    const int t = dir ? (Tn - 2 - s) : (1 + s);

    // x_t A-fragment: two b128 loads per lane (16-bit A layout).
    const unsigned short* xrow = X + ((size_t)t * Bn + brow0 + (lane & 15)) * 32;
    FragU xf;
    xf.q[0] = *reinterpret_cast<const U4*>(xrow + 8 * half);
    xf.q[1] = *reinterpret_cast<const U4*>(xrow + 16 + 8 * half);

    if (s + 1 < Tn - 1) {
      const int t2 = dir ? (Tn - 3 - s) : (2 + s);
      __builtin_prefetch(X + ((size_t)t2 * Bn + brow0 + (lane & 15)) * 32, 0, 3);
    }

    v8f aA[6], aR2[2];
#pragma unroll
    for (int j = 0; j < 6; ++j) {
      float bv = biasA[j];
#pragma unroll
      for (int r = 0; r < 8; ++r) aA[j][r] = bv;
    }
#pragma unroll
    for (int j = 0; j < 2; ++j) {
      float bv = biasR2[j];
#pragma unroll
      for (int r = 0; r < 8; ++r) aR2[j][r] = bv;
    }

    // gi = x_t @ W (+bias), tiles 0..5
#pragma unroll
    for (int j = 0; j < 6; ++j)
      aA[j] = __builtin_amdgcn_wmma_f32_16x16x32_bf16(false, xf.v, false, Wf[j],
                                                      (short)0, aA[j], false, false);
    // z/r gates share accumulators with gr tiles 0..3
#pragma unroll
    for (int j = 0; j < 4; ++j)
      aA[j] = __builtin_amdgcn_wmma_f32_16x16x32_bf16(false, hfrag, false, Uf[j],
                                                      (short)0, aA[j], false, false);
    // recurrent h-gate term kept separate (tiles 4,5 of gr)
    aR2[0] = __builtin_amdgcn_wmma_f32_16x16x32_bf16(false, hfrag, false, Uf[4],
                                                     (short)0, aR2[0], false, false);
    aR2[1] = __builtin_amdgcn_wmma_f32_16x16x32_bf16(false, hfrag, false, Uf[5],
                                                     (short)0, aR2[1], false, false);

    // Gate math + state update (C layout: m = r + 8*half, u = 16*j + n).
    float* outbase = out + (size_t)t * Bn * 32;
#pragma unroll
    for (int j = 0; j < 2; ++j) {
#pragma unroll
      for (int r = 0; r < 8; ++r) {
        float z  = fsigmoid(aA[j][r]);
        float rr = fsigmoid(aA[j + 2][r]);
        float hp = aA[j + 4][r] + rr * aR2[j][r];
        float hh = hp > 0.0f ? hp : 0.0f;
        float hn = z * hstate[j][r] + (1.0f - z) * hh;
        hstate[j][r] = hn;
        int m = r + 8 * half;
        int u = 16 * j + n;
        outbase[(size_t)(brow0 + m) * 32 + u] = hn;
        hs[m][u] = f2bf(hn);
      }
    }
    __syncthreads();  // single wave: NOP per ISA; keeps LDS ordering explicit
    // Rebuild h A-fragment from LDS (ds b128 loads, 16-bit A layout).
    {
      const unsigned short* hrow = &hs[lane & 15][0];
      FragU hf;
      hf.q[0] = *reinterpret_cast<const U4*>(hrow + 8 * half);
      hf.q[1] = *reinterpret_cast<const U4*>(hrow + 16 + 8 * half);
      hfrag = hf.v;
    }
    __syncthreads();
  }
}

// ---------------------------------------------------------------------------
// K4: BN stats over (T,B) per channel: sum & sumsq of merged = outL + outR
// ---------------------------------------------------------------------------
__global__ void bn_stats(const float* __restrict__ a, const float* __restrict__ b,
                         float* __restrict__ stats) {
  __shared__ float ssum[32], ssq[32];
  if (threadIdx.x < 32) { ssum[threadIdx.x] = 0.0f; ssq[threadIdx.x] = 0.0f; }
  __syncthreads();
  const int total = Tn * Bn * 32;
  const int stride = gridDim.x * blockDim.x;  // multiple of 32 -> channel fixed per thread
  int i0 = blockIdx.x * blockDim.x + threadIdx.x;
  int c = i0 & 31;
  float s = 0.0f, q = 0.0f;
  for (int i = i0; i < total; i += stride) {
    float v = a[i] + b[i];
    s += v; q += v * v;
  }
  atomicAdd(&ssum[c], s);
  atomicAdd(&ssq[c], q);
  __syncthreads();
  if (threadIdx.x < 32) {
    atomicAdd(&stats[threadIdx.x], ssum[threadIdx.x]);
    atomicAdd(&stats[32 + threadIdx.x], ssq[threadIdx.x]);
  }
}

// ---------------------------------------------------------------------------
// K5: BN apply -> next layer input X (bf16, time-major)
// ---------------------------------------------------------------------------
__global__ void bn_apply(const float* __restrict__ a, const float* __restrict__ b,
                         const float* __restrict__ stats,
                         const float* __restrict__ gamma, const float* __restrict__ beta,
                         unsigned short* __restrict__ X) {
  const int total = Tn * Bn * 32;
  const int stride = gridDim.x * blockDim.x;
  int i0 = blockIdx.x * blockDim.x + threadIdx.x;
  int c = i0 & 31;
  const float invN = 1.0f / (float)(Tn * Bn);
  float m = stats[c] * invN;
  float var = stats[32 + c] * invN - m * m;
  float sc = gamma[c] * rsqrtf(var + 1e-3f);
  float sh = beta[c] - m * sc;
  for (int i = i0; i < total; i += stride) {
    X[i] = f2bf((a[i] + b[i]) * sc + sh);
  }
}

// ---------------------------------------------------------------------------
// K6: dense (last one only survives): o = relu(X @ Wd + bd), per-(t,c) stats
// ---------------------------------------------------------------------------
__global__ __launch_bounds__(32) void dense_kernel(
    const unsigned short* __restrict__ X, const float* __restrict__ Wd,
    const float* __restrict__ bd, float* __restrict__ o, float* __restrict__ dstats) {
  const int lane = threadIdx.x;
  const int n = lane & 15, half = lane >> 4;
  const int t = blockIdx.y;
  const int brow0 = blockIdx.x * 16;

  v16bf Bf[2];
  Bf[0] = make_bfrag(Wd, 32, 0, lane);
  Bf[1] = make_bfrag(Wd, 32, 16, lane);

  const unsigned short* xrow = X + ((size_t)t * Bn + brow0 + n) * 32;
  FragU xf;
  xf.q[0] = *reinterpret_cast<const U4*>(xrow + 8 * half);
  xf.q[1] = *reinterpret_cast<const U4*>(xrow + 16 + 8 * half);

  v8f acc[2];
#pragma unroll
  for (int j = 0; j < 2; ++j) {
    float bv = bd[16 * j + n];
#pragma unroll
    for (int r = 0; r < 8; ++r) acc[j][r] = bv;
    acc[j] = __builtin_amdgcn_wmma_f32_16x16x32_bf16(false, xf.v, false, Bf[j],
                                                     (short)0, acc[j], false, false);
  }

#pragma unroll
  for (int j = 0; j < 2; ++j) {
    int u = 16 * j + n;
    float s = 0.0f, q = 0.0f;
#pragma unroll
    for (int r = 0; r < 8; ++r) {
      float v = acc[j][r] > 0.0f ? acc[j][r] : 0.0f;
      int m = r + 8 * half;
      o[((size_t)t * Bn + brow0 + m) * 32 + u] = v;
      s += v; q += v * v;
    }
    atomicAdd(&dstats[t * 32 + u], s);
    atomicAdd(&dstats[Tn * 32 + t * 32 + u], q);
  }
}

// ---------------------------------------------------------------------------
// K7: per-timestep BN + 32->11 projection + softmax -> d_out (B,T,11)
// ---------------------------------------------------------------------------
__global__ void final_kernel(const float* __restrict__ o, const float* __restrict__ dstats,
                             const float* __restrict__ gamma, const float* __restrict__ beta,
                             const float* __restrict__ Wf, const float* __restrict__ bf,
                             float* __restrict__ out) {
  int row = blockIdx.x * blockDim.x + threadIdx.x;
  if (row >= Tn * Bn) return;
  int t = row / Bn, b = row % Bn;
  const float* op = o + (size_t)row * 32;
  const float invB = 1.0f / (float)Bn;
  float v[32];
#pragma unroll
  for (int c = 0; c < 32; ++c) {
    float m = dstats[t * 32 + c] * invB;
    float var = dstats[Tn * 32 + t * 32 + c] * invB - m * m;
    v[c] = gamma[c] * (op[c] - m) * rsqrtf(var + 1e-3f) + beta[c];
  }
  float logits[NCn];
  float mx = -3.4e38f;
#pragma unroll
  for (int k = 0; k < NCn; ++k) {
    float a = bf[k];
#pragma unroll
    for (int c = 0; c < 32; ++c) a += v[c] * Wf[c * NCn + k];
    logits[k] = a;
    mx = a > mx ? a : mx;
  }
  float denom = 0.0f;
#pragma unroll
  for (int k = 0; k < NCn; ++k) {
    float e = __expf(logits[k] - mx);
    logits[k] = e;
    denom += e;
  }
  float inv = __builtin_amdgcn_rcpf(denom);
  float* dst = out + ((size_t)b * Tn + t) * NCn;
#pragma unroll
  for (int k = 0; k < NCn; ++k) dst[k] = logits[k] * inv;
}

// ---------------------------------------------------------------------------
extern "C" void kernel_launch(void* const* d_in, const int* in_sizes, int n_in,
                              void* d_out, int out_size, void* d_ws, size_t ws_size,
                              hipStream_t stream) {
  const float* x   = (const float*)d_in[0];
  const float* emb = (const float*)d_in[1];
  const float* Wl  = (const float*)d_in[2];
  const float* Ul  = (const float*)d_in[3];
  const float* bl  = (const float*)d_in[4];
  const float* Wr  = (const float*)d_in[5];
  const float* Ur  = (const float*)d_in[6];
  const float* br  = (const float*)d_in[7];
  const float* bng = (const float*)d_in[8];
  const float* bnb = (const float*)d_in[9];
  const float* Wd  = (const float*)d_in[10];
  const float* bd  = (const float*)d_in[11];
  const float* dg  = (const float*)d_in[12];
  const float* db  = (const float*)d_in[13];
  const float* Wf  = (const float*)d_in[14];
  const float* bf  = (const float*)d_in[15];
  float* out = (float*)d_out;

  // Workspace layout (~80.2 MB): X bf16 | outL f32 | outR f32 | stats | dstats.
  char* w = (char*)d_ws;
  unsigned short* X = (unsigned short*)w;                    // 16 MB
  float* outL  = (float*)(w + ((size_t)16u << 20));          // 32 MB
  float* outR  = (float*)(w + ((size_t)48u << 20));          // 32 MB
  float* stats = (float*)(w + ((size_t)80u << 20));          // 64 floats
  float* dstats = stats + 64;                                // T*32*2 floats
  float* obuf = outL;                                        // reuse after layer loop

  argmax_embed<<<(Bn * Tn * 32 + 255) / 256, 256, 0, stream>>>(x, emb, X);

  for (int i = 0; i < 2; ++i) {
    // boundary slices never written by the GRUs (t=0 left, t=T-1 right)
    zero_f32<<<(Bn * 32 + 255) / 256, 256, 0, stream>>>(outL, Bn * 32);
    zero_f32<<<(Bn * 32 + 255) / 256, 256, 0, stream>>>(outR + (size_t)(Tn - 1) * Bn * 32, Bn * 32);
    zero_f32<<<1, 64, 0, stream>>>(stats, 64);
    gru_kernel<<<dim3(Bn / 16, 2), 32, 0, stream>>>(
        X, outL, outR,
        Wl + (size_t)i * EMBn * 96, Ul + (size_t)i * Un * 96, bl + (size_t)i * 2 * 96,
        Wr + (size_t)i * EMBn * 96, Ur + (size_t)i * Un * 96, br + (size_t)i * 2 * 96);
    bn_stats<<<1024, 256, 0, stream>>>(outL, outR, stats);
    bn_apply<<<2048, 256, 0, stream>>>(outL, outR, stats, bng + i * Un, bnb + i * Un, X);
  }

  zero_f32<<<(Tn * 32 * 2 + 255) / 256, 256, 0, stream>>>(dstats, Tn * 32 * 2);
  dense_kernel<<<dim3(Bn / 16, Tn), 32, 0, stream>>>(X, Wd + 32 * 32, bd + 32, obuf, dstats);
  final_kernel<<<(Tn * Bn + 255) / 256, 256, 0, stream>>>(obuf, dstats, dg + 32, db + 32, Wf, bf, out);
}
